// GATLayer_55499567399479
// MI455X (gfx1250) — matrix operand order repaired and checked by reading, not compiled, not run on hardware
//
#include <hip/hip_runtime.h>
#include <hip/hip_bf16.h>
#include <math.h>

#define IN_DIM    128
#define HEADS     4
#define OUT_DIM   32
#define HD        (HEADS * OUT_DIM)   // 128
#define LEAKY     0.2f

typedef __attribute__((ext_vector_type(16))) __bf16 v16bf;
typedef __attribute__((ext_vector_type(8)))  __bf16 v8bf;
typedef __attribute__((ext_vector_type(4)))  __bf16 v4bf;
typedef __attribute__((ext_vector_type(2)))  __bf16 v2bf;
typedef __attribute__((ext_vector_type(8)))  float  v8f;

__device__ __forceinline__ __bf16 cvt_bf16_sw(float f) {
  // round-to-nearest-even fp32 -> bf16 (software fallback)
  unsigned u = __float_as_uint(f);
  u += 0x7FFFu + ((u >> 16) & 1u);
  unsigned short h = (unsigned short)(u >> 16);
  return __builtin_bit_cast(__bf16, h);
}

__device__ __forceinline__ v2bf cvt_pk_bf16(float lo, float hi) {
#if __has_builtin(__builtin_amdgcn_cvt_pk_bf16_f32)
  return __builtin_amdgcn_cvt_pk_bf16_f32(lo, hi);
#else
  v2bf r;
  r[0] = cvt_bf16_sw(lo);
  r[1] = cvt_bf16_sw(hi);
  return r;
#endif
}

__device__ __forceinline__ float lrelu(float x) {
  return x > 0.f ? x : LEAKY * x;
}

__device__ __forceinline__ void atomicMaxFloat(float* addr, float val) {
  // sign-split trick: works for mixed-sign values when initialized to -inf bits
  if (val >= 0.f) atomicMax((int*)addr, __float_as_int(val));
  else            atomicMin((unsigned int*)addr, __float_as_uint(val));
}

// ---------------------------------------------------------------------------
// Init: out = 0, m = -inf, denom = 0   (harness poisons buffers with 0xAA)
// ---------------------------------------------------------------------------
__global__ __launch_bounds__(256) void gat_init(float* __restrict__ out,
                                                float* __restrict__ mmax,
                                                float* __restrict__ denom,
                                                int n_out, int n_nh) {
  int t = blockIdx.x * blockDim.x + threadIdx.x;
  if (t < n_out) out[t] = 0.f;
  if (t < n_nh) {
    mmax[t]  = __uint_as_float(0xFF800000u); // -inf
    denom[t] = 0.f;
  }
}

// ---------------------------------------------------------------------------
// One-shot fp32 -> bf16 conversion (bandwidth bound; 4 elements / thread).
// Amortizes the RNE conversion once instead of per-wave in the GEMM.
// ---------------------------------------------------------------------------
__global__ __launch_bounds__(256) void cvt_f32_to_bf16(
    const float* __restrict__ in, __bf16* __restrict__ out, int n) {
  int i = (blockIdx.x * blockDim.x + threadIdx.x) * 4;
  if (i >= n) return;
  float4 f = *(const float4*)(in + i);
  v2bf lo = cvt_pk_bf16(f.x, f.y);
  v2bf hi = cvt_pk_bf16(f.z, f.w);
  v4bf o;
  o[0] = lo[0]; o[1] = lo[1]; o[2] = hi[0]; o[3] = hi[1];
  *(v4bf*)(out + i) = o;
}

// ---------------------------------------------------------------------------
// WMMA GEMM: ft[n, j] = sum_k feature[n, k] * W[j, k]   (ft = feature @ W^T)
// One block = 16 nodes; 8 waves, each owns a 16-wide column tile (8*16 = 128).
// K = 128 -> 4 x v_wmma_f32_16x16x32_bf16 per wave. EXEC all-1s (no guards;
// N_NODES % 16 == 0). bf16 operands pre-converted, so per-lane fragments are
// contiguous: A = two b128 loads, B = two b128 loads per K-step. No VALU.
// ---------------------------------------------------------------------------
__global__ __launch_bounds__(256) void gat_gemm_wmma(
    const __bf16* __restrict__ featbf, const __bf16* __restrict__ Wbf,
    float* __restrict__ ft) {
  const int lane   = threadIdx.x & 31;
  const int wave   = threadIdx.x >> 5;     // column tile id, 0..7
  const int node0  = blockIdx.x * 16;
  const int col0   = wave * 16;
  const int l16    = lane & 15;
  const int hi     = lane >> 4;            // 0 or 1 (half-wave)
  const int khalfA = hi * 8;               // A: K sub-offset per ISA layout
  const int kbaseB = hi * 16;              // unused; B is loaded with khalfA, same k order as A

  const __bf16* __restrict__ arow = featbf + (size_t)(node0 + l16) * IN_DIM;
  const __bf16* __restrict__ brow = Wbf    + (size_t)(col0 + l16) * IN_DIM; // B[k][j]=W[j][k]

  v8f c = {};
  #pragma unroll
  for (int kt = 0; kt < 4; ++kt) {
    const int k0 = kt * 32;
    v16bf a, b;
    // A elements 0..7  <- K = [k0+8*hi,    k0+8*hi+8)   (16B aligned)
    // A elements 8..15 <- K = [k0+16+8*hi, k0+16+8*hi+8)
    *(v8bf*)&a         = *(const v8bf*)(arow + k0 + khalfA);
    *(((v8bf*)&a) + 1) = *(const v8bf*)(arow + k0 + khalfA + 16);
    // B elements 0..7 <- K = [k0+8*hi, k0+8*hi+8), 8..15 <- K = [k0+16+8*hi, k0+16+8*hi+8)
    *(v8bf*)&b         = *(const v8bf*)(brow + k0 + khalfA);
    *(((v8bf*)&b) + 1) = *(const v8bf*)(brow + k0 + khalfA + 16);
    c = __builtin_amdgcn_wmma_f32_16x16x32_bf16(false, a, false, b,
                                                (short)0, c, false, false);
  }

  // C/D layout: VGPR r -> row = r + 8*hi, col = lane&15
  #pragma unroll
  for (int r = 0; r < 8; ++r) {
    int row = node0 + r + 8 * hi;
    ft[(size_t)row * HD + col0 + l16] = c[r];
  }
}

// ---------------------------------------------------------------------------
// Per-node attention logits: a1[n,h] = dot(ft[n,h,:], attn_l[h,:]), same for a2
// ---------------------------------------------------------------------------
__global__ __launch_bounds__(256) void gat_logits(
    const float* __restrict__ ft, const float* __restrict__ attn_l,
    const float* __restrict__ attn_r, float* __restrict__ a1,
    float* __restrict__ a2, int n_nodes) {
  int t = blockIdx.x * blockDim.x + threadIdx.x;
  if (t >= n_nodes * HEADS) return;
  int n = t >> 2, h = t & 3;
  const float* f  = ft + (size_t)n * HD + h * OUT_DIM;
  const float* al = attn_l + h * OUT_DIM;
  const float* ar = attn_r + h * OUT_DIM;
  float s1 = 0.f, s2 = 0.f;
  #pragma unroll
  for (int d = 0; d < OUT_DIM; ++d) {
    float v = f[d];
    s1 += v * al[d];
    s2 += v * ar[d];
  }
  a1[t] = s1;
  a2[t] = s2;
}

// ---------------------------------------------------------------------------
// Pass 1 over edges: m[dst,h] = max over edges of leaky_relu(a1[src]+a2[dst])
// ---------------------------------------------------------------------------
__global__ __launch_bounds__(256) void gat_edge_max(
    const float* __restrict__ a1, const float* __restrict__ a2,
    const int* __restrict__ src, const int* __restrict__ dst,
    float* __restrict__ mmax, int n_edges) {
  int t = blockIdx.x * blockDim.x + threadIdx.x;
  int e = t >> 2, h = t & 3;
  if (e >= n_edges) return;
  int s = src[e], d = dst[e];
  float v = lrelu(a1[s * HEADS + h] + a2[d * HEADS + h]);
  atomicMaxFloat(&mmax[d * HEADS + h], v);
}

// ---------------------------------------------------------------------------
// Pass 2: denom[dst,h] += exp(e - m[dst,h])
// ---------------------------------------------------------------------------
__global__ __launch_bounds__(256) void gat_edge_sum(
    const float* __restrict__ a1, const float* __restrict__ a2,
    const int* __restrict__ src, const int* __restrict__ dst,
    const float* __restrict__ mmax, float* __restrict__ denom, int n_edges) {
  int t = blockIdx.x * blockDim.x + threadIdx.x;
  int e = t >> 2, h = t & 3;
  if (e >= n_edges) return;
  int s = src[e], d = dst[e];
  float v  = lrelu(a1[s * HEADS + h] + a2[d * HEADS + h]);
  float ex = __expf(v - mmax[d * HEADS + h]);
  atomicAdd(&denom[d * HEADS + h], ex);
}

// ---------------------------------------------------------------------------
// Pass 3: out[dst, h, :] += alpha * ft[src, h, :]
// One wave32 per edge: lane = feature dim, loop over 4 heads.
// ft and out are L2-resident (25.6 MB each within 192 MB), so the random
// gather and the global_atomic_add_f32 scatter resolve on-chip.
// ---------------------------------------------------------------------------
__global__ __launch_bounds__(256) void gat_aggregate(
    const float* __restrict__ ft, const float* __restrict__ a1,
    const float* __restrict__ a2, const float* __restrict__ mmax,
    const float* __restrict__ denom, const int* __restrict__ src,
    const int* __restrict__ dst, float* __restrict__ out, int n_edges) {
  int wid  = (blockIdx.x * blockDim.x + threadIdx.x) >> 5;
  int lane = threadIdx.x & 31;
  if (wid >= n_edges) return;
  int s = src[wid], d = dst[wid];
  const float* __restrict__ fs = ft + (size_t)s * HD;
  float* __restrict__ od = out + (size_t)d * HD;
  #pragma unroll
  for (int h = 0; h < HEADS; ++h) {
    float e     = lrelu(a1[s * HEADS + h] + a2[d * HEADS + h]);
    float alpha = __expf(e - mmax[d * HEADS + h]) / denom[d * HEADS + h];
    atomicAdd(&od[h * OUT_DIM + lane], alpha * fs[h * OUT_DIM + lane]);
  }
}

// ---------------------------------------------------------------------------
extern "C" void kernel_launch(void* const* d_in, const int* in_sizes, int n_in,
                              void* d_out, int out_size, void* d_ws, size_t ws_size,
                              hipStream_t stream) {
  const float* feature = (const float*)d_in[0];   // [N, 128]
  const float* W       = (const float*)d_in[1];   // [128, 128]
  const float* attn_l  = (const float*)d_in[2];   // [1, 4, 32]
  const float* attn_r  = (const float*)d_in[3];   // [1, 4, 32]
  const int*   src     = (const int*)d_in[4];     // [E]
  const int*   dst     = (const int*)d_in[5];     // [E]

  const int n_nodes = in_sizes[0] / IN_DIM;       // 50000
  const int n_edges = in_sizes[4];                // 1,600,000
  const int n_feat  = n_nodes * IN_DIM;           // 6.4M
  const int n_w     = HD * IN_DIM;                // 16384

  float* out = (float*)d_out;                     // [N, 4, 32] = [N, 128]

  // Workspace layout: ft | a1 | a2 | m | denom (floats) | featbf | Wbf (bf16)
  // All section offsets are 32B aligned.
  float*  ws     = (float*)d_ws;
  float*  ft     = ws;
  float*  a1     = ft + (size_t)n_nodes * HD;
  float*  a2     = a1 + (size_t)n_nodes * HEADS;
  float*  mmax   = a2 + (size_t)n_nodes * HEADS;
  float*  denom  = mmax + (size_t)n_nodes * HEADS;
  __bf16* featbf = (__bf16*)(denom + (size_t)n_nodes * HEADS);
  __bf16* Wbf    = featbf + (size_t)n_feat;

  const int n_out = n_nodes * HD;                 // 6.4M
  const int n_nh  = n_nodes * HEADS;              // 200K

  // 1. zero out / init softmax accumulators
  gat_init<<<(n_out + 255) / 256, 256, 0, stream>>>(out, mmax, denom, n_out, n_nh);

  // 2. one-shot bf16 conversion of feature and W
  cvt_f32_to_bf16<<<(n_feat / 4 + 255) / 256, 256, 0, stream>>>(feature, featbf, n_feat);
  cvt_f32_to_bf16<<<(n_w / 4 + 255) / 256, 256, 0, stream>>>(W, Wbf, n_w);

  // 3. ft = feature @ W^T via WMMA (bf16 in, fp32 acc)
  gat_gemm_wmma<<<n_nodes / 16, 256, 0, stream>>>(featbf, Wbf, ft);

  // 4. per-node logits
  gat_logits<<<(n_nh + 255) / 256, 256, 0, stream>>>(ft, attn_l, attn_r, a1, a2, n_nodes);

  // 5. segment max over incoming edges
  gat_edge_max<<<(n_edges * HEADS + 255) / 256, 256, 0, stream>>>(
      a1, a2, src, dst, mmax, n_edges);

  // 6. segment sum of exp(e - m)
  gat_edge_sum<<<(n_edges * HEADS + 255) / 256, 256, 0, stream>>>(
      a1, a2, src, dst, mmax, denom, n_edges);

  // 7. weighted scatter: out[dst] += alpha * ft[src]  (one wave per edge)
  long long agg_threads = (long long)n_edges * 32;
  gat_aggregate<<<(unsigned)((agg_threads + 255) / 256), 256, 0, stream>>>(
      ft, a1, a2, mmax, denom, src, dst, out, n_edges);
}
